// GroupQueryAttention_24541443129579
// MI455X (gfx1250) — compile-verified, matrix operand order checked by reading, not run on hardware
//
#include <hip/hip_runtime.h>
#include <math.h>

// ---------------------------------------------------------------------------
// GQA forward for MI455X (gfx1250), bf16 WMMA (v_wmma_f32_16x16x32_bf16).
// Group-sum-before-softmax quirk folded into Wq (Wq_eff, 2048x512).
// Pipeline: prep weights -> convert Q/K/V to bf16 -> 3 WMMA GEMM projections
// (double-buffered async global->LDS tiles) -> causal flash attention (WMMA,
// async K/V staging) -> WMMA out-projection.
// ---------------------------------------------------------------------------

typedef __attribute__((ext_vector_type(16))) __bf16 v16bf;
typedef __attribute__((ext_vector_type(8)))  float  v8f;

union AFrag {
  v16bf v;
  uint4 q[2];      // two 16B chunks: K = half*8..+7 and 16+half*8..+7
};

#define B_   2
#define S_   2048
#define D_   2048
#define HKV  4
#define DK   128
#define DHID 512   // HKV*DK

// ---------------- async global->LDS helpers (CDNA5 ASYNCcnt path) ----------
#if __has_builtin(__builtin_amdgcn_global_load_async_to_lds_b128)
#define HAVE_ASYNC_LDS 1
#endif

typedef int v4i_vec __attribute__((vector_size(16)));
typedef __attribute__((address_space(1))) v4i_vec* glob_v4i;
typedef __attribute__((address_space(3))) v4i_vec* lds_v4i;

__device__ __forceinline__ void async_ld16(void* lds, const void* gl)
{
#ifdef HAVE_ASYNC_LDS
  // generic LDS pointer: low 32 bits are the LDS byte offset (ISA aperture rule)
  __builtin_amdgcn_global_load_async_to_lds_b128(
      (glob_v4i)(uintptr_t)gl,
      (lds_v4i)(uint32_t)(uintptr_t)lds,
      0, 0);
#else
  *reinterpret_cast<uint4*>(lds) = *reinterpret_cast<const uint4*>(gl);
#endif
}

#ifdef HAVE_ASYNC_LDS
#if __has_builtin(__builtin_amdgcn_s_wait_asynccnt)
#define WAIT_ASYNC(n) __builtin_amdgcn_s_wait_asynccnt(n)
#else
#define WAIT_ASYNC(n) asm volatile("s_wait_asynccnt %0" ::"n"(n) : "memory")
#endif
#else
#define WAIT_ASYNC(n) ((void)0)
#endif

// ---------------- weight prep: fold group-sum, transpose, bf16 -------------
__global__ __launch_bounds__(256)
void prep_weights(const float* __restrict__ Wq, const float* __restrict__ Wk,
                  const float* __restrict__ Wv, const float* __restrict__ Wo,
                  __bf16* __restrict__ WqT,   // [512][2048]  (N x K), scaled
                  __bf16* __restrict__ WkT,   // [512][2048]
                  __bf16* __restrict__ WvT,   // [512][2048]
                  __bf16* __restrict__ WoT)   // [2048][512]
{
  int idx = blockIdx.x * blockDim.x + threadIdx.x;
  const int NK = DHID * D_;
  const float scale = 0.0883883476483184f;   // 1/sqrt(128) folded into q
  if (idx < NK) {
    int n = idx / D_, k = idx % D_;
    int h = n >> 7, dk = n & 127;
    float s = 0.f;
#pragma unroll
    for (int g = 0; g < 4; ++g)
      s += Wq[(size_t)k * D_ + (g * HKV + h) * DK + dk];
    WqT[idx] = (__bf16)(s * scale);
  } else if (idx < 2 * NK) {
    int t = idx - NK; int n = t / D_, k = t % D_;
    WkT[t] = (__bf16)Wk[(size_t)k * DHID + n];
  } else if (idx < 3 * NK) {
    int t = idx - 2 * NK; int n = t / D_, k = t % D_;
    WvT[t] = (__bf16)Wv[(size_t)k * DHID + n];
  } else if (idx < 4 * NK) {
    int t = idx - 3 * NK; int n = t / DHID, k = t % DHID;
    WoT[t] = (__bf16)Wo[(size_t)k * D_ + n];
  }
}

// ---------------- fp32 -> bf16 activation convert --------------------------
__global__ __launch_bounds__(256)
void convert_bf16(const float4* __restrict__ in, __bf16* __restrict__ out, int n4)
{
  int i = blockIdx.x * blockDim.x + threadIdx.x;
  if (i >= n4) return;
  float4 f = in[i];
  union { __bf16 b[4]; uint2 u; } p;
  p.b[0] = (__bf16)f.x; p.b[1] = (__bf16)f.y;
  p.b[2] = (__bf16)f.z; p.b[3] = (__bf16)f.w;
  reinterpret_cast<uint2*>(out)[i] = p.u;
}

// ---------------- tiled bf16 WMMA GEMM:  C = A[MxK] * (BT[NxK])^T ----------
// 256 threads = 8 waves, 128x128 C tile, K-step 32. Wave = 32x64 sub-tile.
// Double-buffered async global->LDS tile loads (4 async b128 per thread/tile).
// MODE 0: bf16 out row-major.  MODE 1: bf16 out scattered as vT[b][h][dv][lq].
// MODE 2: f32 out row-major.
#define GT_K 32
#define APAD 8      // LDS row stride 40 bf16 (80B) -> conflict-free b128 reads

template <int MODE>
__global__ __launch_bounds__(256)
void gemm_bf16(const __bf16* __restrict__ A, const __bf16* __restrict__ BT,
               void* __restrict__ Cout, int M, int N, int K)
{
  __shared__ __bf16 As[2][128][GT_K + APAD];
  __shared__ __bf16 Bs[2][128][GT_K + APAD];

  const int tid  = threadIdx.x;
  const int lane = tid & 31, wave = tid >> 5;
  const int hi   = lane >> 4, lnm = lane & 15;
  const int m0 = blockIdx.y * 128;
  const int n0 = blockIdx.x * 128;
  const int wm = (wave & 3) * 32;   // 2 M-frags
  const int wn = (wave >> 2) * 64;  // 4 N-frags

  v8f acc[2][4];
#pragma unroll
  for (int i = 0; i < 2; ++i)
#pragma unroll
    for (int j = 0; j < 4; ++j)
#pragma unroll
      for (int r = 0; r < 8; ++r) acc[i][j][r] = 0.f;

  const int lrow = tid >> 1;            // 0..127
  const int lcol = (tid & 1) * 16;      // bf16 units
  const __bf16* gA = A  + (size_t)(m0 + lrow) * K + lcol;
  const __bf16* gB = BT + (size_t)(n0 + lrow) * K + lcol;

  auto issue = [&](int k0, int buf) {
    async_ld16(&As[buf][lrow][lcol],     gA + k0);
    async_ld16(&As[buf][lrow][lcol + 8], gA + k0 + 8);
    async_ld16(&Bs[buf][lrow][lcol],     gB + k0);
    async_ld16(&Bs[buf][lrow][lcol + 8], gB + k0 + 8);
  };

  issue(0, 0);
  const int T = K / GT_K;
  for (int t = 0; t < T; ++t) {
    const int cur = t & 1;
    if (t + 1 < T) {
      issue((t + 1) * GT_K, cur ^ 1);   // prefetch next tile (4 newest ops)
      WAIT_ASYNC(4);                    // in-order: oldest 4 (this tile) done
    } else {
      WAIT_ASYNC(0);
    }
    __syncthreads();

    AFrag af[2], bf[4];
#pragma unroll
    for (int i = 0; i < 2; ++i) {
      int row = wm + i * 16 + lnm;
      af[i].q[0] = *reinterpret_cast<const uint4*>(&As[cur][row][hi * 8]);
      af[i].q[1] = *reinterpret_cast<const uint4*>(&As[cur][row][16 + hi * 8]);
    }
#pragma unroll
    for (int j = 0; j < 4; ++j) {
      int row = wn + j * 16 + lnm;
      bf[j].q[0] = *reinterpret_cast<const uint4*>(&Bs[cur][row][hi * 8]);
      bf[j].q[1] = *reinterpret_cast<const uint4*>(&Bs[cur][row][16 + hi * 8]);
    }
#pragma unroll
    for (int i = 0; i < 2; ++i)
#pragma unroll
      for (int j = 0; j < 4; ++j)
        acc[i][j] = __builtin_amdgcn_wmma_f32_16x16x32_bf16(
            false, af[i].v, false, bf[j].v, (short)0, acc[i][j], false, false);
    __syncthreads();   // all waves done reading buf[cur] before it is refilled
  }

#pragma unroll
  for (int i = 0; i < 2; ++i)
#pragma unroll
    for (int j = 0; j < 4; ++j)
#pragma unroll
      for (int r = 0; r < 8; ++r) {
        int row = m0 + wm + i * 16 + r + 8 * hi;   // C layout: M = r + 8*(lane/16)
        int col = n0 + wn + j * 16 + lnm;          //           N = lane%16
        float v = acc[i][j][r];
        if (MODE == 0) {
          ((__bf16*)Cout)[(size_t)row * N + col] = (__bf16)v;
        } else if (MODE == 1) {
          int b = row >> 11, lq = row & 2047;
          int h = col >> 7,  dv = col & 127;
          ((__bf16*)Cout)[(((size_t)(b * HKV + h) * DK + dv) << 11) + lq] = (__bf16)v;
        } else {
          ((float*)Cout)[(size_t)row * N + col] = v;
        }
      }
}

// ---------------- causal flash attention (per b,h) -------------------------
// Block: 256 thr = 8 waves; 128 q-rows per block (16 per wave); 64-key tiles.
#define FQ  128
#define FKT 64
#define KP  8

__global__ __launch_bounds__(256)
void flash_attn(const __bf16* __restrict__ Qp,  // [B*S][512]
                const __bf16* __restrict__ Kp,  // [B*S][512]
                const __bf16* __restrict__ VT,  // [B*H][128][S]
                __bf16* __restrict__ Ob)        // [B*S][512]
{
  __shared__ __bf16 Ks[FKT][DK + KP];           // 64 x 136
  __shared__ __bf16 Vs[DK][FKT + KP];           // 128 x 72
  __shared__ __bf16 Ps[8][16][FKT + KP];        // per-wave P relayout buffer

  const int tid  = threadIdx.x;
  const int lane = tid & 31, wave = tid >> 5;
  const int hi   = lane >> 4, lnm = lane & 15;
  const int bh = blockIdx.y;                    // b*HKV + h
  const int b  = bh >> 2, h = bh & 3;
  const int q0 = blockIdx.x * FQ;
  const int qrow = q0 + wave * 16 + lnm;        // this lane's A-frag row

  // q fragments: 16 rows x 128 dk = 4 K-steps, held in registers
  AFrag qf[4];
  const __bf16* qbase = Qp + (size_t)(b * S_ + qrow) * DHID + h * DK;
#pragma unroll
  for (int kk = 0; kk < 4; ++kk) {
    qf[kk].q[0] = *reinterpret_cast<const uint4*>(qbase + kk * 32 + hi * 8);
    qf[kk].q[1] = *reinterpret_cast<const uint4*>(qbase + kk * 32 + 16 + hi * 8);
  }

  v8f oacc[8];
#pragma unroll
  for (int i = 0; i < 8; ++i)
#pragma unroll
    for (int r = 0; r < 8; ++r) oacc[i][r] = 0.f;
  float mrow[8], lrow[8], alpha[8];
#pragma unroll
  for (int r = 0; r < 8; ++r) { mrow[r] = -1e30f; lrow[r] = 0.f; }

  // per-thread staging coordinates
  const int krow = tid >> 2, kpart = (tid & 3) * 32;   // K tile: 4 thr/row
  const int vdv  = tid >> 1, vpart = (tid & 1) * 32;   // V tile: 2 thr/row
  const __bf16* kbase = Kp + (size_t)(b * S_ + krow) * DHID + h * DK + kpart;
  const __bf16* vbase = VT + ((size_t)bh * DK + vdv) * (size_t)S_ + vpart;

  const int ntiles = (q0 + FQ) / FKT;           // covers all j <= q0+127
  for (int t = 0; t < ntiles; ++t) {
    const int j0 = t * FKT;
    {   // async-stage K tile [64 keys][128 dk] and V^T tile [128 dv][64 keys]
      const __bf16* src = kbase + (size_t)j0 * DHID;
      async_ld16(&Ks[krow][kpart],      src);
      async_ld16(&Ks[krow][kpart + 8],  src + 8);
      async_ld16(&Ks[krow][kpart + 16], src + 16);
      async_ld16(&Ks[krow][kpart + 24], src + 24);
      const __bf16* vsrc = vbase + j0;
      async_ld16(&Vs[vdv][vpart],      vsrc);
      async_ld16(&Vs[vdv][vpart + 8],  vsrc + 8);
      async_ld16(&Vs[vdv][vpart + 16], vsrc + 16);
      async_ld16(&Vs[vdv][vpart + 24], vsrc + 24);
      WAIT_ASYNC(0);
    }
    __syncthreads();

    // S = q * k^T : 4 N-frags (16 keys) x 4 K-steps (dk)
    v8f sacc[4];
#pragma unroll
    for (int j = 0; j < 4; ++j)
#pragma unroll
      for (int r = 0; r < 8; ++r) sacc[j][r] = 0.f;
#pragma unroll
    for (int kk = 0; kk < 4; ++kk)
#pragma unroll
      for (int j = 0; j < 4; ++j) {
        AFrag kb;
        int kr = j * 16 + lnm;
        kb.q[0] = *reinterpret_cast<const uint4*>(&Ks[kr][kk * 32 + hi * 8]);
        kb.q[1] = *reinterpret_cast<const uint4*>(&Ks[kr][kk * 32 + 16 + hi * 8]);
        sacc[j] = __builtin_amdgcn_wmma_f32_16x16x32_bf16(
            false, qf[kk].v, false, kb.v, (short)0, sacc[j], false, false);
      }

    // causal mask + online softmax (row's 16 cols live in lanes 0..15 of half)
#pragma unroll
    for (int r = 0; r < 8; ++r) {
      int qr = q0 + wave * 16 + r + 8 * hi;
      float mx = -1e30f;
#pragma unroll
      for (int j = 0; j < 4; ++j) {
        int col = j0 + j * 16 + lnm;
        float sv = (col <= qr) ? sacc[j][r] : -1e30f;
        sacc[j][r] = sv;
        mx = fmaxf(mx, sv);
      }
#pragma unroll
      for (int off = 1; off < 16; off <<= 1)
        mx = fmaxf(mx, __shfl_xor(mx, off, 32));
      float mnew = fmaxf(mrow[r], mx);
      float a = __expf(mrow[r] - mnew);
      mrow[r] = mnew; alpha[r] = a;
      float ls = 0.f;
#pragma unroll
      for (int j = 0; j < 4; ++j) {
        float p = __expf(sacc[j][r] - mnew);
        sacc[j][r] = p;
        ls += p;
      }
#pragma unroll
      for (int off = 1; off < 16; off <<= 1)
        ls += __shfl_xor(ls, off, 32);
      lrow[r] = lrow[r] * a + ls;
    }

    // relayout P (C layout -> A layout) through per-wave LDS
#pragma unroll
    for (int j = 0; j < 4; ++j)
#pragma unroll
      for (int r = 0; r < 8; ++r)
        Ps[wave][r + 8 * hi][j * 16 + lnm] = (__bf16)sacc[j][r];

#pragma unroll
    for (int of = 0; of < 8; ++of)
#pragma unroll
      for (int r = 0; r < 8; ++r) oacc[of][r] *= alpha[r];

    // O += P * V : 2 K-steps (keys) x 8 N-frags (dv)
#pragma unroll
    for (int kk = 0; kk < 2; ++kk) {
      AFrag pa;
      pa.q[0] = *reinterpret_cast<const uint4*>(&Ps[wave][lnm][kk * 32 + hi * 8]);
      pa.q[1] = *reinterpret_cast<const uint4*>(&Ps[wave][lnm][kk * 32 + 16 + hi * 8]);
#pragma unroll
      for (int of = 0; of < 8; ++of) {
        AFrag vb;
        int dvr = of * 16 + lnm;
        vb.q[0] = *reinterpret_cast<const uint4*>(&Vs[dvr][kk * 32 + hi * 8]);
        vb.q[1] = *reinterpret_cast<const uint4*>(&Vs[dvr][kk * 32 + 16 + hi * 8]);
        oacc[of] = __builtin_amdgcn_wmma_f32_16x16x32_bf16(
            false, pa.v, false, vb.v, (short)0, oacc[of], false, false);
      }
    }
    __syncthreads();
  }

  // normalize and store O as bf16 [B*S][512]
#pragma unroll
  for (int r = 0; r < 8; ++r) {
    float inv = 1.f / lrow[r];
    int row = q0 + wave * 16 + r + 8 * hi;
    __bf16* ob = Ob + (size_t)(b * S_ + row) * DHID + h * DK;
#pragma unroll
    for (int of = 0; of < 8; ++of)
      ob[of * 16 + lnm] = (__bf16)(oacc[of][r] * inv);
  }
}

// ---------------------------------------------------------------------------
extern "C" void kernel_launch(void* const* d_in, const int* in_sizes, int n_in,
                              void* d_out, int out_size, void* d_ws, size_t ws_size,
                              hipStream_t stream)
{
  const float* Q  = (const float*)d_in[0];
  const float* K  = (const float*)d_in[1];
  const float* V  = (const float*)d_in[2];
  const float* Wq = (const float*)d_in[3];
  const float* Wk = (const float*)d_in[4];
  const float* Wv = (const float*)d_in[5];
  const float* Wo = (const float*)d_in[6];
  float* out = (float*)d_out;

  char* ws = (char*)d_ws;
  auto alloc = [&](size_t bytes) {
    char* p = ws; ws += (bytes + 255) & ~(size_t)255; return p;
  };
  const size_t BS = (size_t)B_ * S_;
  __bf16* Qbf = (__bf16*)alloc(BS * D_ * 2);      // 16 MB
  __bf16* Kbf = (__bf16*)alloc(BS * D_ * 2);
  __bf16* Vbf = (__bf16*)alloc(BS * D_ * 2);
  __bf16* WqT = (__bf16*)alloc((size_t)DHID * D_ * 2);
  __bf16* WkT = (__bf16*)alloc((size_t)DHID * D_ * 2);
  __bf16* WvT = (__bf16*)alloc((size_t)DHID * D_ * 2);
  __bf16* WoT = (__bf16*)alloc((size_t)D_ * DHID * 2);
  __bf16* qp  = (__bf16*)alloc(BS * DHID * 2);
  __bf16* kp  = (__bf16*)alloc(BS * DHID * 2);
  __bf16* vT  = (__bf16*)alloc(BS * DHID * 2);
  __bf16* Ob  = (__bf16*)alloc(BS * DHID * 2);    // total ~72 MB

  { // weights: fold group-sum into Wq, transpose all to [N][K] bf16
    int total = 4 * DHID * D_;
    prep_weights<<<dim3((total + 255) / 256), dim3(256), 0, stream>>>(
        Wq, Wk, Wv, Wo, WqT, WkT, WvT, WoT);
  }
  { // activations -> bf16
    int n4 = (int)(BS * D_ / 4);
    dim3 g((n4 + 255) / 256), blk(256);
    convert_bf16<<<g, blk, 0, stream>>>((const float4*)Q, Qbf, n4);
    convert_bf16<<<g, blk, 0, stream>>>((const float4*)K, Kbf, n4);
    convert_bf16<<<g, blk, 0, stream>>>((const float4*)V, Vbf, n4);
  }
  { // q/k/v projections (4096x2048 @ 2048x512); v written transposed
    dim3 grid(DHID / 128, (int)(BS / 128)), blk(256);
    gemm_bf16<0><<<grid, blk, 0, stream>>>(Qbf, WqT, qp, (int)BS, DHID, D_);
    gemm_bf16<0><<<grid, blk, 0, stream>>>(Kbf, WkT, kp, (int)BS, DHID, D_);
    gemm_bf16<1><<<grid, blk, 0, stream>>>(Vbf, WvT, vT, (int)BS, DHID, D_);
  }
  { // causal flash attention per (b, h)
    dim3 grid(S_ / FQ, B_ * HKV), blk(256);
    flash_attn<<<grid, blk, 0, stream>>>(qp, kp, vT, Ob);
  }
  { // output projection (4096x512 @ 512x2048), f32 out
    dim3 grid(D_ / 128, (int)(BS / 128)), blk(256);
    gemm_bf16<2><<<grid, blk, 0, stream>>>(Ob, WoT, out, (int)BS, D_, DHID);
  }
}
